// GCN_77653008712280
// MI455X (gfx1250) — matrix operand order repaired and checked by reading, not compiled, not run on hardware
//
#include <hip/hip_runtime.h>

// ---------------------------------------------------------------------------
// 2-layer GCN on MI455X (gfx1250, wave32, WMMA).
//
// Roofline: GEMMs total ~3.3 GFLOP (trivial for v_wmma_f32_16x16x32_f16).
// The aggregation moves ~2.6 GB of edge gather traffic -> ~115us floor at
// 23.3 TB/s HBM. Naive per-feature float atomics would be ~307M
// global_atomic_add_f32 ops and bottleneck the L2 atomic units, so we build
// a CSR (column-sorted) adjacency in-launch (int histogram + LDS prefix scan
// + 1.6M int placement atomics) and aggregate atomic-free: per target node a
// wave does coalesced gathers of xw[row,:] and a single plain store.
// Weights (32KB), dis[] (400KB) and edge_src[] (6.4MB) are L2-resident.
// ReLU is fused into the layer-2 WMMA A-operand conversion.
// ---------------------------------------------------------------------------

typedef __attribute__((ext_vector_type(16))) _Float16 v16h;
typedef __attribute__((ext_vector_type(8)))  float    v8f;

// ---- CSR build: histogram ---------------------------------------------------

__global__ void zero_int_kernel(int* __restrict__ p, int n) {
  int i = blockIdx.x * blockDim.x + threadIdx.x;
  if (i < n) p[i] = 0;
}

__global__ void count_edges_kernel(const long long* __restrict__ col,
                                   int* __restrict__ cnt, int e) {
  int i = blockIdx.x * blockDim.x + threadIdx.x;
  if (i < e) atomicAdd(&cnt[(int)col[i]], 1);
}

// dis[i] = 1/sqrt(deg) with deg = in-edge count + 1 (self-loop); always > 0.
__global__ void inv_sqrt_kernel(const int* __restrict__ cnt,
                                float* __restrict__ dis, int n) {
  int i = blockIdx.x * blockDim.x + threadIdx.x;
  if (i < n) dis[i] = rsqrtf((float)cnt[i] + 1.0f);
}

// ---- CSR build: exclusive prefix scan (1024/block Hillis-Steele in LDS) -----

__global__ void scan_block_kernel(const int* __restrict__ cnt,
                                  int* __restrict__ starts,
                                  int* __restrict__ blockSums, int n) {
  __shared__ int s[1024];
  const int tid = threadIdx.x;
  const int gid = blockIdx.x * 1024 + tid;
  int v = (gid < n) ? cnt[gid] : 0;
  s[tid] = v;
  __syncthreads();
#pragma unroll
  for (int off = 1; off < 1024; off <<= 1) {
    int t = (tid >= off) ? s[tid - off] : 0;
    __syncthreads();
    s[tid] += t;
    __syncthreads();
  }
  if (gid < n) starts[gid] = s[tid] - v;            // exclusive
  if (tid == 1023) blockSums[blockIdx.x] = s[tid];  // block total
}

__global__ void scan_sums_kernel(int* __restrict__ blockSums, int nb) {
  __shared__ int s[1024];
  const int tid = threadIdx.x;
  int v = (tid < nb) ? blockSums[tid] : 0;
  s[tid] = v;
  __syncthreads();
#pragma unroll
  for (int off = 1; off < 1024; off <<= 1) {
    int t = (tid >= off) ? s[tid - off] : 0;
    __syncthreads();
    s[tid] += t;
    __syncthreads();
  }
  if (tid < nb) blockSums[tid] = s[tid] - v;        // exclusive block offsets
}

__global__ void add_offsets_kernel(int* __restrict__ starts,
                                   const int* __restrict__ blockSums, int n) {
  int gid = blockIdx.x * blockDim.x + threadIdx.x;
  if (gid < n) starts[gid] += blockSums[gid >> 10];
}

// ---- CSR build: placement (only atomics left: 1.6M int adds) ----------------

__global__ void place_edges_kernel(const long long* __restrict__ erow,
                                   const long long* __restrict__ ecol,
                                   const int* __restrict__ starts,
                                   int* __restrict__ cursor,
                                   int* __restrict__ edge_src, int e) {
  int i = blockIdx.x * blockDim.x + threadIdx.x;
  if (i < e) {
    int c = (int)ecol[i];
    int pos = starts[c] + atomicAdd(&cursor[c], 1);
    edge_src[pos] = (int)erow[i];
  }
}

// ---- WMMA GEMM: Y[n,FOUT] = (RELU? max(X,0):X)[n,FIN] @ W[FIN,FOUT] ---------
// One wave per 16x16 output tile. K-chunks of 32.
//
// ISA 7.12.2 layouts (wave32):
//  A 16x32 f16: lane<16 -> row M=lane,     halves 0..7 = K 0..7,  8..15 = K 16..23
//               lane>=16 -> row M=lane-16, halves 0..7 = K 8..15, 8..15 = K 24..31
//  B 32x16 f16: lane<16 -> col N=lane,     halves 0..7 = K 0..7,  8..15 = K 16..23
//               lane>=16 -> col N=lane-16, halves 0..7 = K 8..15, 8..15 = K 24..31
//  C/D 16x16 f32: vgpr r, lane l -> element (M = r + 8*(l>=16), N = l%16)

template<int FIN, int FOUT, bool RELU>
__global__ void gemm_wmma_kernel(const float* __restrict__ X,
                                 const float* __restrict__ W,
                                 float* __restrict__ Y,
                                 int tiles_total) {
  const int lane = threadIdx.x & 31;
  const int wave = threadIdx.x >> 5;
  const int tile = blockIdx.x * (blockDim.x >> 5) + wave;
  if (tile >= tiles_total) return;            // wave-uniform: EXEC stays all-1s
  constexpr int TILES_N = FOUT / 16;
  const int tM   = tile / TILES_N;
  const int tN   = tile % TILES_N;
  const int lmod = lane & 15;
  const int hi   = lane >> 4;                 // 0 = lanes 0-15, 1 = lanes 16-31
  const int mrow = tM * 16 + lmod;
  const int ncol = tN * 16 + lmod;

  v8f acc = {};
#pragma unroll
  for (int kk = 0; kk < FIN; kk += 32) {
    v16h a, b;
    // A operand: 8+8 contiguous fp32 -> f16 (b128 loads + v_cvt_pk_f16_f32)
    const float* xr = X + (size_t)mrow * FIN + kk + hi * 8;
#pragma unroll
    for (int i = 0; i < 8; ++i) {
      float v0 = xr[i];
      float v1 = xr[16 + i];
      if (RELU) { v0 = fmaxf(v0, 0.0f); v1 = fmaxf(v1, 0.0f); }
      a[i]     = (_Float16)v0;
      a[8 + i] = (_Float16)v1;
    }
    // B operand: column ncol, K rows 8h..8h+7 and 16+8h..16+8h+7 (stride FOUT; L2-resident)
    const float* wr = W + (size_t)(kk + hi * 8) * FOUT + ncol;
#pragma unroll
    for (int i = 0; i < 16; ++i) b[i] = (_Float16)wr[(size_t)(i + (i & 8)) * FOUT];

    acc = __builtin_amdgcn_wmma_f32_16x16x32_f16(
        /*neg_a=*/false, a, /*neg_b=*/false, b,
        /*c_mod=*/(short)0, acc, /*reuse_a=*/false, /*reuse_b=*/false);
  }

  float* yr = Y + (size_t)(tM * 16 + hi * 8) * FOUT + ncol;
#pragma unroll
  for (int r = 0; r < 8; ++r) yr[(size_t)r * FOUT] = acc[r];
}

// ---- atomic-free CSR aggregation --------------------------------------------
// out[c,f] = bias[f] + dis[c]^2 * xw[c,f] + sum_{j in CSR(c)} dis[r_j]*dis[c]*xw[r_j,f]
// One wave per (node, 32-feature chunk). Lanes cooperatively load a batch of
// 32 edge records (coalesced), then broadcast each edge via __shfl so the
// xw[row, fbase..fbase+31] gather is one coalesced 128B line per edge.

template<int F>
__global__ void aggregate_kernel(const float* __restrict__ xw,
                                 const int* __restrict__ starts,
                                 const int* __restrict__ cnt,
                                 const int* __restrict__ edge_src,
                                 const float* __restrict__ dis,
                                 const float* __restrict__ bias,
                                 float* __restrict__ out, int n) {
  constexpr int CHUNKS = F / 32;
  const int lane = threadIdx.x & 31;
  const int wave = threadIdx.x >> 5;
  const int gw   = blockIdx.x * (blockDim.x >> 5) + wave;
  const int node = gw / CHUNKS;
  const int chnk = gw % CHUNKS;
  if (node >= n) return;
  const int f = chnk * 32 + lane;

  const float dc = dis[node];
  // self-loop (norm = dis[c]^2) + bias
  float acc = xw[(size_t)node * F + f] * (dc * dc) + bias[f];

  const int s  = starts[node];
  const int eN = s + cnt[node];
  for (int j0 = s; j0 < eN; j0 += 32) {
    const int myIdx = j0 + lane;
    int   rL = 0;
    float wL = 0.0f;
    if (myIdx < eN) {
      rL = edge_src[myIdx];          // coalesced int load
      wL = dis[rL] * dc;             // edge norm
    }
    const int m = min(32, eN - j0);
    for (int t = 0; t < m; ++t) {
      int   r = __shfl(rL, t, 32);   // broadcast edge to all lanes
      float w = __shfl(wL, t, 32);
      acc += xw[(size_t)r * F + f] * w;   // coalesced 128B gather
    }
  }
  out[(size_t)node * F + f] = acc;        // plain store, no atomics
}

// ---------------------------------------------------------------------------

extern "C" void kernel_launch(void* const* d_in, const int* in_sizes, int n_in,
                              void* d_out, int out_size, void* d_ws, size_t ws_size,
                              hipStream_t stream) {
  (void)n_in; (void)out_size; (void)ws_size;

  const float*     x    = (const float*)d_in[0];
  const long long* eidx = (const long long*)d_in[1];   // int64 [2, E]
  const float*     W1   = (const float*)d_in[2];       // [64, 128]
  const float*     b1   = (const float*)d_in[3];       // [128]
  const float*     W2   = (const float*)d_in[4];       // [128, 64]
  const float*     b2   = (const float*)d_in[5];       // [64]
  float*           out  = (float*)d_out;               // [N, 64]

  const int n = in_sizes[0] / 64;     // 100000 (6250 row tiles of 16)
  const int e = in_sizes[1] / 2;      // 1600000
  const long long* erow = eidx;
  const long long* ecol = eidx + e;
  const int nb = (n + 1023) / 1024;   // scan blocks (98)

  // Workspace layout: [cnt n][cursor n][starts n][blockSums 1024][edge_src e]
  //                   [dis n floats][bufA n*128 f][bufH n*128 f]  (~110 MB)
  int*   cnt     = (int*)d_ws;
  int*   cursor  = cnt + n;
  int*   starts  = cursor + n;
  int*   bsum    = starts + n;
  int*   edgesrc = bsum + 1024;
  float* dis     = (float*)(edgesrc + e);
  float* bufA    = dis + n;                    // xw1, reused for xw2
  float* bufH    = bufA + (size_t)n * 128;     // layer-1 activations (pre-ReLU)

  constexpr int T = 256;   // 8 wave32 per block

  // --- CSR build + norms ---
  zero_int_kernel<<<(2 * n + T - 1) / T, T, 0, stream>>>(cnt, 2 * n);  // cnt+cursor
  count_edges_kernel<<<(e + T - 1) / T, T, 0, stream>>>(ecol, cnt, e);
  inv_sqrt_kernel<<<(n + T - 1) / T, T, 0, stream>>>(cnt, dis, n);
  scan_block_kernel<<<nb, 1024, 0, stream>>>(cnt, starts, bsum, n);
  scan_sums_kernel<<<1, 1024, 0, stream>>>(bsum, nb);
  add_offsets_kernel<<<(n + T - 1) / T, T, 0, stream>>>(starts, bsum, n);
  place_edges_kernel<<<(e + T - 1) / T, T, 0, stream>>>(erow, ecol, starts, cursor,
                                                        edgesrc, e);

  // --- layer 1: xw1 = x @ W1 ; h = aggregate(xw1) + b1 ---
  {
    int tiles  = (n / 16) * (128 / 16);
    gemm_wmma_kernel<64, 128, false><<<(tiles + 7) / 8, T, 0, stream>>>(x, W1, bufA, tiles);
  }
  {
    int waves  = n * (128 / 32);
    aggregate_kernel<128><<<(waves + 7) / 8, T, 0, stream>>>(bufA, starts, cnt, edgesrc,
                                                             dis, b1, bufH, n);
  }

  // --- layer 2: xw2 = relu(h) @ W2 ; out = aggregate(xw2) + b2 ---
  {
    int tiles  = (n / 16) * (64 / 16);
    gemm_wmma_kernel<128, 64, true><<<(tiles + 7) / 8, T, 0, stream>>>(bufH, W2, bufA, tiles);
  }
  {
    int waves  = n * (64 / 32);
    aggregate_kernel<64><<<(waves + 7) / 8, T, 0, stream>>>(bufA, starts, cnt, edgesrc,
                                                            dis, b2, out, n);
  }
}
